// CompILE_10127532884476
// MI455X (gfx1250) — matrix operand order repaired and checked by reading, not compiled, not run on hardware
//
#include <hip/hip_runtime.h>
#include <hip/hip_bf16.h>
#include <math.h>

typedef __bf16 bf16;
typedef __attribute__((ext_vector_type(16))) __bf16 bf16x16;
typedef __attribute__((ext_vector_type(8)))  float  f32x8;

#define BB  32
#define TT  512
#define SS  128
#define AA  32
#define HH  256
#define KK  16
#define EMB 512   // 2H
#define GG  1024  // 4H
#define BT  (BB*TT)

// ---------------- WMMA fragment helpers (bf16, 16x16x32, row-major sources) ----------------
// A: 16xK tile, row-major source with leading dim ld.
// Lanes 0-15 hold row=lane, K elems {k0..k0+7, k0+16..k0+23}; lanes 16-31 row=lane-16, +8.
__device__ __forceinline__ bf16x16 load_a_bf16(const bf16* base, int ld, int k0) {
  int lane = threadIdx.x & 31;
  int row  = lane & 15;
  int kb   = k0 + ((lane >> 4) << 3);
  const bf16* p = base + row * ld + kb;
  union { uint4 u[2]; bf16x16 v; } t;
  t.u[0] = *(const uint4*)(p);
  t.u[1] = *(const uint4*)(p + 16);
  return t.v;
}
// B: K x 16 tile where B[k][n] = W[n][k], W row-major [N][Kdim].
// Lanes 0-15: col n0+lane, k = {k0..k0+7, k0+16..k0+23}; lanes 16-31: col n0+lane-16, +8.
__device__ __forceinline__ bf16x16 load_b_w(const bf16* W, int ldk, int n0, int k0) {
  int lane = threadIdx.x & 31;
  int n    = n0 + (lane & 15);
  int kb   = k0 + ((lane >> 4) << 3);
  const bf16* p = W + n * ldk + kb;
  union { uint4 u[2]; bf16x16 v; } t;
  t.u[0] = *(const uint4*)(p);
  t.u[1] = *(const uint4*)(p + 16);
  return t.v;
}
__device__ __forceinline__ f32x8 wmma_bf16(bf16x16 a, bf16x16 b, f32x8 c) {
  return __builtin_amdgcn_wmma_f32_16x16x32_bf16(false, a, false, b, (short)0, c, false, false);
}
__device__ __forceinline__ unsigned hash_u32(unsigned x) {
  x ^= x >> 16; x *= 0x7feb352dU; x ^= x >> 15; x *= 0x846ca68bU; x ^= x >> 16;
  return x;
}
__device__ __forceinline__ float sigf(float x) { return 1.f / (1.f + expf(-x)); }

// ---------------- conversion / layout kernels ----------------
__global__ void k_f32_to_bf16(const float* __restrict__ src, bf16* __restrict__ dst, int n) {
  int i = blockIdx.x * blockDim.x + threadIdx.x;
  if (i < n) dst[i] = (bf16)src[i];
}
// Wsub [K][A][S] -> [a*K+k][S] bf16 (flatten btak einsum to a GEMM with N = A*K)
__global__ void k_wsub_reorder(const float* __restrict__ Wsub, bf16* __restrict__ dst) {
  int i = blockIdx.x * blockDim.x + threadIdx.x;
  if (i >= AA * KK * SS) return;
  int s = i % SS; int n = i / SS; int k = n % KK; int a = n / KK;
  dst[n * SS + s] = (bf16)Wsub[(k * AA + a) * SS + s];
}
// action embedding gather into emb[:, H:2H]
__global__ void k_gather_act(const int* __restrict__ actions,
                             const float* __restrict__ emb_action,
                             bf16* __restrict__ emb) {
  int i = blockIdx.x * blockDim.x + threadIdx.x;
  if (i >= BT * HH) return;
  int r = i / HH, j = i % HH;
  emb[r * EMB + HH + j] = (bf16)emb_action[actions[r] * HH + j];
}
__global__ void k_init(float* __restrict__ logcum, float* __restrict__ maskbuf) {
  int i = blockIdx.x * blockDim.x + threadIdx.x;
  if (i < BT) { logcum[i] = 0.f; maskbuf[i] = 1.f; }
}

// ---------------- se = relu(states@Ws1T+bs1)@Ws2T+bs2 -> emb[:, 0:H] ----------------
__global__ __launch_bounds__(128) void k_se(const float* __restrict__ states,
    const bf16* __restrict__ Ws1, const float* __restrict__ bs1,
    const bf16* __restrict__ Ws2, const float* __restrict__ bs2,
    bf16* __restrict__ emb) {
  __shared__ __align__(16) bf16 st[16 * SS];
  __shared__ __align__(16) bf16 hid[16 * HH];
  int tid = threadIdx.x;
  int rbase = blockIdx.x * 16;
  for (int i = tid; i < 16 * SS; i += 128)
    st[i] = (bf16)states[(size_t)(rbase + i / SS) * SS + (i % SS)];
  __syncthreads();
  int w = tid >> 5, lane = tid & 31;
  int col = lane & 15, rhi = (lane >> 4) << 3;
  for (int nt = w * 4; nt < w * 4 + 4; ++nt) {
    int n0 = nt * 16;
    f32x8 acc = {};
    for (int kt = 0; kt < SS / 32; ++kt)
      acc = wmma_bf16(load_a_bf16(st, SS, kt * 32), load_b_w(Ws1, SS, n0, kt * 32), acc);
    int n = n0 + col;
    float bias = bs1[n];
#pragma unroll
    for (int v = 0; v < 8; ++v) {
      float x = acc[v] + bias;
      hid[(v + rhi) * HH + n] = (bf16)(x > 0.f ? x : 0.f);
    }
  }
  __syncthreads();
  for (int nt = w * 4; nt < w * 4 + 4; ++nt) {
    int n0 = nt * 16;
    f32x8 acc = {};
    for (int kt = 0; kt < HH / 32; ++kt)
      acc = wmma_bf16(load_a_bf16(hid, HH, kt * 32), load_b_w(Ws2, HH, n0, kt * 32), acc);
    int n = n0 + col;
    float bias = bs2[n];
#pragma unroll
    for (int v = 0; v < 8; ++v)
      emb[(size_t)(rbase + v + rhi) * EMB + n] = (bf16)(acc[v] + bias);
  }
}

// ---------------- Xg = emb@WihT + bih + bhh  (segment-invariant LSTM input gates) ----------------
__global__ __launch_bounds__(128) void k_xgates(const bf16* __restrict__ emb,
    const bf16* __restrict__ Wih, const float* __restrict__ bih,
    const float* __restrict__ bhh, float* __restrict__ Xg) {
  int w = threadIdx.x >> 5, lane = threadIdx.x & 31;
  int rbase = blockIdx.x * 16;
  int n0 = blockIdx.y * 64 + w * 16;
  f32x8 acc = {};
  const bf16* arow = emb + (size_t)rbase * EMB;
  for (int kt = 0; kt < EMB / 32; ++kt)
    acc = wmma_bf16(load_a_bf16(arow, EMB, kt * 32), load_b_w(Wih, EMB, n0, kt * 32), acc);
  int col = lane & 15, rhi = (lane >> 4) << 3;
  int n = n0 + col;
  float bias = bih[n] + bhh[n];
#pragma unroll
  for (int v = 0; v < 8; ++v)
    Xg[(size_t)(rbase + v + rhi) * GG + n] = acc[v] + bias;
}

// ---------------- sub_logits GEMM (N = A*K) then softmax over A ----------------
__global__ __launch_bounds__(128) void k_sublogits(const bf16* __restrict__ st,
    const bf16* __restrict__ Wsub, const float* __restrict__ bsub,
    float* __restrict__ out) {
  int w = threadIdx.x >> 5, lane = threadIdx.x & 31;
  int rbase = blockIdx.x * 16;
  int n0 = blockIdx.y * 64 + w * 16;
  f32x8 acc = {};
  const bf16* arow = st + (size_t)rbase * SS;
  for (int kt = 0; kt < SS / 32; ++kt)
    acc = wmma_bf16(load_a_bf16(arow, SS, kt * 32), load_b_w(Wsub, SS, n0, kt * 32), acc);
  int col = lane & 15, rhi = (lane >> 4) << 3;
  int n = n0 + col;
  float bias = bsub[(n % KK) * AA + (n / KK)];
#pragma unroll
  for (int v = 0; v < 8; ++v)
    out[(size_t)(rbase + v + rhi) * (AA * KK) + n] = acc[v] + bias;
}
__global__ void k_softmax_a(float* __restrict__ sp) {
  int i = blockIdx.x * blockDim.x + threadIdx.x;   // over BT*KK
  if (i >= BT * KK) return;
  int bt = i / KK, k = i % KK;
  float* p = sp + (size_t)bt * (AA * KK) + k;
  float m = -1e30f;
  for (int a = 0; a < AA; ++a) m = fmaxf(m, p[a * KK]);
  float s = 0.f;
  for (int a = 0; a < AA; ++a) s += expf(p[a * KK] - m);
  float inv = 1.f / s;
  for (int a = 0; a < AA; ++a) p[a * KK] = expf(p[a * KK] - m) * inv;
}

// ---------------- persistent masked-LSTM recurrence: one workgroup, 32 waves ----------------
// wave w: mt=w&1 picks 16 batch rows, ng=w>>1 picks 16 hidden cols; the wave computes
// its i/f/g/o tiles (cols j, H+j, 2H+j, 3H+j) so the pointwise update is wave-local.
__global__ __launch_bounds__(1024) void k_lstm(const float* __restrict__ Xg,
    const bf16* __restrict__ Whh, const float* __restrict__ maskbuf,
    float* __restrict__ enc) {
  __shared__ __align__(16) bf16 hsh[BB * HH];
  int tid = threadIdx.x;
  for (int i = tid; i < BB * HH; i += 1024) hsh[i] = (bf16)0.f;
  int w = tid >> 5, lane = tid & 31;
  int mt = w & 1, ng = w >> 1;
  int n0 = ng * 16;
  int col = lane & 15, rhi = (lane >> 4) << 3;
  int j = n0 + col;
  float c[8];
#pragma unroll
  for (int v = 0; v < 8; ++v) c[v] = 0.f;
  __syncthreads();
  for (int t = 0; t < TT; ++t) {
    f32x8 ai = {}, af = {}, ag = {}, ao = {};
    for (int kt = 0; kt < HH / 32; ++kt) {
      bf16x16 a = load_a_bf16(hsh + mt * 16 * HH, HH, kt * 32);
      ai = wmma_bf16(a, load_b_w(Whh, HH, 0 * HH + n0, kt * 32), ai);
      af = wmma_bf16(a, load_b_w(Whh, HH, 1 * HH + n0, kt * 32), af);
      ag = wmma_bf16(a, load_b_w(Whh, HH, 2 * HH + n0, kt * 32), ag);
      ao = wmma_bf16(a, load_b_w(Whh, HH, 3 * HH + n0, kt * 32), ao);
    }
    float hnew[8];
#pragma unroll
    for (int v = 0; v < 8; ++v) {
      int m = mt * 16 + v + rhi;
      const float* xg = Xg + (size_t)(m * TT + t) * GG;
      float gi = ai[v] + xg[j];
      float gf = af[v] + xg[HH + j];
      float gg = ag[v] + xg[2 * HH + j];
      float go = ao[v] + xg[3 * HH + j];
      float cn = sigf(gf) * c[v] + sigf(gi) * tanhf(gg);
      float hn = sigf(go) * tanhf(cn);
      float mk = maskbuf[m * TT + t];
      hn *= mk; cn *= mk;
      c[v] = cn; hnew[v] = hn;
      enc[((size_t)m * TT + t) * HH + j] = hn;
    }
    __syncthreads();
#pragma unroll
    for (int v = 0; v < 8; ++v)
      hsh[(mt * 16 + v + rhi) * HH + j] = (bf16)hnew[v];
    __syncthreads();
  }
}

// ---------------- boundary logits: relu(enc@Wb1T+bb1)@Wb2 + bb2, t==0 -> NEG_INF ----------------
__global__ __launch_bounds__(128) void k_blogits(const float* __restrict__ enc,
    const bf16* __restrict__ Wb1, const float* __restrict__ bb1,
    const float* __restrict__ Wb2, const float* __restrict__ bb2,
    float* __restrict__ out) {
  __shared__ __align__(16) bf16 eb[16 * HH];
  __shared__ float partial[16];
  int tid = threadIdx.x;
  int rbase = blockIdx.x * 16;
  for (int i = tid; i < 16 * HH; i += 128)
    eb[i] = (bf16)enc[(size_t)(rbase + i / HH) * HH + (i % HH)];
  if (tid < 16) partial[tid] = 0.f;
  __syncthreads();
  int w = tid >> 5, lane = tid & 31;
  int col = lane & 15, rhi = (lane >> 4) << 3;
  for (int nt = w * 4; nt < w * 4 + 4; ++nt) {
    int n0 = nt * 16;
    f32x8 acc = {};
    for (int kt = 0; kt < HH / 32; ++kt)
      acc = wmma_bf16(load_a_bf16(eb, HH, kt * 32), load_b_w(Wb1, HH, n0, kt * 32), acc);
    int n = n0 + col;
    float b1 = bb1[n], w2 = Wb2[n];
#pragma unroll
    for (int v = 0; v < 8; ++v) {
      float x = acc[v] + b1;
      x = x > 0.f ? x : 0.f;
      atomicAdd(&partial[v + rhi], x * w2);
    }
  }
  __syncthreads();
  if (tid < 16) {
    int bt = rbase + tid;
    int t = bt % TT;
    out[bt] = (t == 0) ? -1e30f : (partial[tid] + bb2[0]);
  }
}

// ---------------- gumbel-softmax sample over T, cumsum, mask update ----------------
__global__ __launch_bounds__(512) void k_sampleb(const float* __restrict__ logits,
    const int* __restrict__ lengths, float* __restrict__ sample_out,
    float* __restrict__ logcum, float* __restrict__ maskbuf,
    float* __restrict__ masks_out, int seg) {
  __shared__ float sh[TT];
  __shared__ float red[16];
  int b = blockIdx.x, t = threadIdx.x;
  if (seg == 3) {
    sample_out[b * TT + t] = (t == lengths[b] - 1) ? 1.f : 0.f;
    return;
  }
  float x = logits[b * TT + t];
  unsigned h = hash_u32(((unsigned)seg * 0x9e3779b9U) ^ ((unsigned)(b * TT + t) * 0x85ebca6bU) ^ 0x1234567U);
  float u = ((float)(h >> 8) + 0.5f) * (1.f / 16777216.f);
  float v = x + (-logf(-logf(u)));          // TEMP_B = 1
  // block max
  float m = v;
#pragma unroll
  for (int off = 16; off; off >>= 1) m = fmaxf(m, __shfl_xor(m, off));
  if ((t & 31) == 0) red[t >> 5] = m;
  __syncthreads();
  if (t == 0) { float mm = red[0]; for (int i = 1; i < 16; ++i) mm = fmaxf(mm, red[i]); red[0] = mm; }
  __syncthreads();
  float M = red[0];
  float e = expf(v - M);
  float s = e;
#pragma unroll
  for (int off = 16; off; off >>= 1) s += __shfl_xor(s, off);
  __syncthreads();
  if ((t & 31) == 0) red[t >> 5] = s;
  __syncthreads();
  if (t == 0) { float a = 0.f; for (int i = 0; i < 16; ++i) a += red[i]; red[0] = a; }
  __syncthreads();
  float p = e / red[0];
  sample_out[b * TT + t] = p;
  sh[t] = p;
  __syncthreads();
  // inclusive scan (Hillis-Steele)
  for (int off = 1; off < TT; off <<= 1) {
    float add = (t >= off) ? sh[t - off] : 0.f;
    __syncthreads();
    sh[t] += add;
    __syncthreads();
  }
  float lc = logcum[b * TT + t] + logf(sh[t] + 1e-17f);
  logcum[b * TT + t] = lc;
  float mk = expf(lc);
  maskbuf[b * TT + t] = mk;
  masks_out[b * TT + t] = mk;
}

// ---------------- readout[b,h] = sum_t enc[b,t,h]*sample_b[b,t+1] ----------------
__global__ __launch_bounds__(256) void k_readout(const float* __restrict__ enc,
    const float* __restrict__ sb, float* __restrict__ ro) {
  __shared__ float sh[TT];
  int b = blockIdx.x, j = threadIdx.x;
  for (int i = j; i < TT; i += 256) sh[i] = sb[b * TT + i];
  __syncthreads();
  float acc = 0.f;
  for (int t = 0; t < TT - 1; ++t)
    acc += enc[((size_t)b * TT + t) * HH + j] * sh[t + 1];
  ro[b * HH + j] = acc;
}

// ---------------- z head: hz=relu(ro@Wz1T+bz1); lz=hz@Wz2T+bz2; reparam sample ----------------
__global__ __launch_bounds__(256) void k_z(const float* __restrict__ ro,
    const float* __restrict__ Wz1, const float* __restrict__ bz1,
    const float* __restrict__ Wz2, const float* __restrict__ bz2,
    float* __restrict__ lz_out, float* __restrict__ sz_out,
    float* __restrict__ zbuf, int seg) {
  __shared__ float hz[BB * HH];
  __shared__ float lz[BB * 2 * KK];
  int tid = threadIdx.x;
  for (int i = tid; i < BB * HH; i += 256) {
    int m = i / HH, n = i % HH;
    float a = bz1[n];
    for (int k = 0; k < HH; ++k) a += ro[m * HH + k] * Wz1[n * HH + k];
    hz[i] = a > 0.f ? a : 0.f;
  }
  __syncthreads();
  for (int i = tid; i < BB * 2 * KK; i += 256) {
    int m = i / (2 * KK), n = i % (2 * KK);
    float a = bz2[n];
    for (int k = 0; k < HH; ++k) a += hz[m * HH + k] * Wz2[n * HH + k];
    lz[i] = a;
    lz_out[i] = a;
  }
  __syncthreads();
  for (int i = tid; i < BB * KK; i += 256) {
    int m = i / KK, k = i % KK;
    float mu = lz[m * 2 * KK + k], lv = lz[m * 2 * KK + KK + k];
    unsigned h1 = hash_u32(0xC0FFEEU ^ ((unsigned)(100 + seg) * 1315423911U) ^ ((unsigned)i * 2654435761U));
    unsigned h2 = hash_u32(h1 ^ 0x9E3779B9U);
    float u1 = ((float)(h1 >> 8) + 0.5f) * (1.f / 16777216.f);
    float u2 = ((float)(h2 >> 8) + 0.5f) * (1.f / 16777216.f);
    float eps = sqrtf(-2.f * logf(u1)) * cosf(6.2831853f * u2);
    float z = mu + eps * expf(0.5f * lv);
    sz_out[i] = z;
    zbuf[i] = z;
  }
}

// ---------------- rec[b,t,a] = sum_k sub_probs[b,t,a,k]*z[b,k] ----------------
__global__ void k_rec(const float* __restrict__ sp, const float* __restrict__ z,
                      float* __restrict__ rec) {
  int i = blockIdx.x * blockDim.x + threadIdx.x;   // BT*AA
  if (i >= BT * AA) return;
  int bt = i / AA, a = i % AA;
  int b = bt / TT;
  const float* p = sp + (size_t)bt * (AA * KK) + a * KK;
  const float* zz = z + b * KK;
  float acc = 0.f;
#pragma unroll
  for (int k = 0; k < KK; ++k) acc += p[k] * zz[k];
  rec[i] = acc;
}

// ---------------- host launch ----------------
extern "C" void kernel_launch(void* const* d_in, const int* in_sizes, int n_in,
                              void* d_out, int out_size, void* d_ws, size_t ws_size,
                              hipStream_t stream) {
  const float* states     = (const float*)d_in[0];
  const int*   actions    = (const int*)  d_in[1];
  const int*   lengths    = (const int*)  d_in[2];
  const float* emb_action = (const float*)d_in[3];
  const float* Ws1 = (const float*)d_in[4];  const float* bs1 = (const float*)d_in[5];
  const float* Ws2 = (const float*)d_in[6];  const float* bs2 = (const float*)d_in[7];
  const float* Wih = (const float*)d_in[8];  const float* bih = (const float*)d_in[9];
  const float* Whh = (const float*)d_in[10]; const float* bhh = (const float*)d_in[11];
  const float* Wz1 = (const float*)d_in[12]; const float* bz1 = (const float*)d_in[13];
  const float* Wz2 = (const float*)d_in[14]; const float* bz2 = (const float*)d_in[15];
  const float* Wb1 = (const float*)d_in[16]; const float* bb1 = (const float*)d_in[17];
  const float* Wb2 = (const float*)d_in[18]; const float* bb2 = (const float*)d_in[19];
  const float* Wsub = (const float*)d_in[20]; const float* bsub = (const float*)d_in[21];

  // workspace carve-up (256B aligned)
  char* ws = (char*)d_ws;
  size_t off = 0;
  auto take = [&](size_t bytes) { char* p = ws + off; off = (off + bytes + 255) & ~(size_t)255; return p; };
  bf16*  emb_bf    = (bf16*) take((size_t)BT * EMB * 2);
  bf16*  states_bf = (bf16*) take((size_t)BT * SS * 2);
  bf16*  Wih_bf    = (bf16*) take((size_t)GG * EMB * 2);
  bf16*  Whh_bf    = (bf16*) take((size_t)GG * HH * 2);
  bf16*  Ws1_bf    = (bf16*) take((size_t)HH * SS * 2);
  bf16*  Ws2_bf    = (bf16*) take((size_t)HH * HH * 2);
  bf16*  Wb1_bf    = (bf16*) take((size_t)HH * HH * 2);
  bf16*  Wsub_bf   = (bf16*) take((size_t)AA * KK * SS * 2);
  float* Xg        = (float*)take((size_t)BT * GG * 4);
  float* sub       = (float*)take((size_t)BT * AA * KK * 4);
  float* logcum    = (float*)take((size_t)BT * 4);
  float* maskbuf   = (float*)take((size_t)BT * 4);
  float* ro        = (float*)take((size_t)BB * HH * 4);
  float* zbuf      = (float*)take((size_t)BB * KK * 4);

  // output layout (flat, return order)
  float* out      = (float*)d_out;
  float* encs     = out;                                   // 4*BT*HH
  float* recs     = encs + (size_t)4 * BT * HH;            // 4*BT*AA
  float* masks    = recs + (size_t)4 * BT * AA;            // 3*BT
  float* logitsb  = masks + (size_t)3 * BT;                // 3*BT
  float* samplesb = logitsb + (size_t)3 * BT;              // 4*BT
  float* logitsz  = samplesb + (size_t)4 * BT;             // 4*BB*2K
  float* samplesz = logitsz + (size_t)4 * BB * 2 * KK;     // 4*BB*K

  // weight / input conversions
  k_f32_to_bf16<<<(BT * SS + 255) / 256, 256, 0, stream>>>(states, states_bf, BT * SS);
  k_f32_to_bf16<<<(HH * SS + 255) / 256, 256, 0, stream>>>(Ws1, Ws1_bf, HH * SS);
  k_f32_to_bf16<<<(HH * HH + 255) / 256, 256, 0, stream>>>(Ws2, Ws2_bf, HH * HH);
  k_f32_to_bf16<<<(GG * EMB + 255) / 256, 256, 0, stream>>>(Wih, Wih_bf, GG * EMB);
  k_f32_to_bf16<<<(GG * HH + 255) / 256, 256, 0, stream>>>(Whh, Whh_bf, GG * HH);
  k_f32_to_bf16<<<(HH * HH + 255) / 256, 256, 0, stream>>>(Wb1, Wb1_bf, HH * HH);
  k_wsub_reorder<<<(AA * KK * SS + 255) / 256, 256, 0, stream>>>(Wsub, Wsub_bf);

  // shared pre-computation
  k_se<<<BT / 16, 128, 0, stream>>>(states, Ws1_bf, bs1, Ws2_bf, bs2, emb_bf);
  k_gather_act<<<(BT * HH + 255) / 256, 256, 0, stream>>>(actions, emb_action, emb_bf);
  k_xgates<<<dim3(BT / 16, GG / 64), 128, 0, stream>>>(emb_bf, Wih_bf, bih, bhh, Xg);
  k_sublogits<<<dim3(BT / 16, (AA * KK) / 64), 128, 0, stream>>>(states_bf, Wsub_bf, bsub, sub);
  k_softmax_a<<<(BT * KK + 255) / 256, 256, 0, stream>>>(sub);
  k_init<<<(BT + 255) / 256, 256, 0, stream>>>(logcum, maskbuf);

  // sequential segments
  for (int seg = 0; seg < 4; ++seg) {
    float* enc_seg = encs + (size_t)seg * BT * HH;
    k_lstm<<<1, 1024, 0, stream>>>(Xg, Whh_bf, maskbuf, enc_seg);
    if (seg < 3)
      k_blogits<<<BT / 16, 128, 0, stream>>>(enc_seg, Wb1_bf, bb1, Wb2, bb2, logitsb + (size_t)seg * BT);
    k_sampleb<<<BB, TT, 0, stream>>>(
        (seg < 3) ? (logitsb + (size_t)seg * BT) : logitsb, lengths,
        samplesb + (size_t)seg * BT, logcum, maskbuf,
        (seg < 3) ? (masks + (size_t)seg * BT) : masks, seg);
    k_readout<<<BB, 256, 0, stream>>>(enc_seg, samplesb + (size_t)seg * BT, ro);
    k_z<<<1, 256, 0, stream>>>(ro, Wz1, bz1, Wz2, bz2,
                               logitsz + (size_t)seg * BB * 2 * KK,
                               samplesz + (size_t)seg * BB * KK, zbuf, seg);
    k_rec<<<(BT * AA + 255) / 256, 256, 0, stream>>>(sub, zbuf, recs + (size_t)seg * BT * AA);
  }
  (void)in_sizes; (void)n_in; (void)out_size; (void)ws_size;
}
